// SIMTransformerBlock_66297115181096
// MI455X (gfx1250) — matrix operand order repaired and checked by reading, not compiled, not run on hardware
//
#include <hip/hip_runtime.h>
#include <hip/hip_bf16.h>

// ---------------------------------------------------------------------------
// CDNA5 (gfx1250) wave32 WMMA implementation of a Swin-style transformer block
// Round 5: single-pass LeFF kernels (X1/G read once from HBM), dwconv tiled
// through LDS (F read ~1.5x instead of 9x), more WMMAs in flight per barrier.
// ---------------------------------------------------------------------------

typedef __attribute__((ext_vector_type(8)))  _Float16 v8h;
typedef __attribute__((ext_vector_type(16))) _Float16 v16h;
typedef __attribute__((ext_vector_type(8)))  float    v8f;

#define DIMC   256
#define WIN    10
#define NH     8
#define HD     32
#define NTOK   100
#define NPAD   112          // 7 * 16
#define HID    1024
#define HH     320
#define WW     320
#define CS     128          // dwconv channel slab
#define SCALE_Q 0.17677669529663687f   // 32^-0.5

__device__ __forceinline__ v8f wmma_f16(v16h a, v16h b, v8f c) {
  return __builtin_amdgcn_wmma_f32_16x16x32_f16(
      /*neg_a=*/false, a, /*neg_b=*/false, b,
      /*c_mod=*/(short)0, c, /*reuse_a=*/false, /*reuse_b=*/false);
}

__device__ __forceinline__ v16h make_v16(v8h lo, v8h hi) {
  v16h r;
#pragma unroll
  for (int e = 0; e < 8; ++e) { r[e] = lo[e]; r[8 + e] = hi[e]; }
  return r;
}

// A fragment (16x32, f16), row-major source with stride lda halves.
__device__ __forceinline__ v16h frag_a_rm(const _Float16* A, int lda) {
  const int lane = threadIdx.x & 31;
  const int m = lane & 15, hf = lane >> 4;
  const _Float16* p = A + m * lda + hf * 16;
  v8h lo = *(const v8h*)p;
  v8h hi = *(const v8h*)(p + 8);
  return make_v16(lo, hi);
}

// B fragment (32x16) from a PRE-PACKED tile: 32 lanes x 16 contiguous halves.
__device__ __forceinline__ v16h frag_b_pk(const _Float16* P) {
  const int lane = threadIdx.x & 31;
  const _Float16* p = P + lane * 16;
  v8h lo = *(const v8h*)p;
  v8h hi = *(const v8h*)(p + 8);
  return make_v16(lo, hi);
}

// B fragment where logical B[k][n] = S[n][k]  (source holds n-major rows of k).
__device__ __forceinline__ v16h frag_b_tr(const _Float16* S, int lds) {
  const int lane = threadIdx.x & 31;
  const int n = lane & 15, hh = lane >> 4;
  const _Float16* p = S + n * lds + hh * 16;
  v8h lo = *(const v8h*)p;
  v8h hi = *(const v8h*)(p + 8);
  return make_v16(lo, hi);
}

__device__ __forceinline__ float gelu_exact(float v) {
  return 0.5f * v * (1.0f + erff(v * 0.70710678118654752f));
}

// ===========================================================================
// Weight packing: fp32 row-major [K x N]  ->  f16 WMMA-B fragment layout.
// ===========================================================================
__global__ __launch_bounds__(256)
void k_pack_b(const float* __restrict__ W, _Float16* __restrict__ P,
              int K, int N) {
  const int idx = blockIdx.x * 256 + threadIdx.x;
  const int ntiles = N >> 4;
  const int total = (K >> 5) * ntiles * 32;
  if (idx >= total) return;
  const int lane = idx & 31;
  const int t = idx >> 5;
  const int nt = t % ntiles, kt = t / ntiles;
  const int n = lane & 15, hh = lane >> 4;
  const float* src = W + (size_t)(kt * 32 + hh * 16) * N + nt * 16 + n;
  _Float16* dst = P + (size_t)idx * 16;
#pragma unroll
  for (int e = 0; e < 16; ++e) dst[e] = (_Float16)src[(size_t)e * N];
}

// ===========================================================================
// Kernel 1: window attention.  One workgroup (8 waves / 256 threads) per
// window; ~308 KB dynamic LDS (CDNA5 WGP has 320 KB).
// ===========================================================================
__global__ __launch_bounds__(256)
void k_win_attn(const float* __restrict__ x,
                const float* __restrict__ g1, const float* __restrict__ b1,
                const _Float16* __restrict__ wqP, const float* __restrict__ bq,
                const _Float16* __restrict__ wkvP, const float* __restrict__ bkv,
                const float* __restrict__ rpb,
                const _Float16* __restrict__ pwP, const float* __restrict__ pb,
                float* __restrict__ X1) {
  extern __shared__ __align__(16) char smem[];
  _Float16* sH  = (_Float16*)smem;             // 112x256  (LN'd input, later O)
  _Float16* sQ  = sH + NPAD * DIMC;            // 112x256
  _Float16* sK  = sQ + NPAD * DIMC;            // 112x256
  _Float16* sVT = sK + NPAD * DIMC;            // 256x128  V TRANSPOSED [chan][tok]
  _Float16* sP  = sVT + DIMC * 128;            // 112x128  (softmax probs, f16)
  float*    sS  = (float*)(sP + NPAD * 128);   // 112x112  (scores, fp32)

  const int tid = threadIdx.x, lane = tid & 31;
  const int wvu = __builtin_amdgcn_readfirstlane(tid >> 5);  // scalar wave id
  const int win = blockIdx.x, wy = win >> 5, wx = win & 31;

  for (int i = tid; i < 12 * DIMC; i += 256) sH[NTOK * DIMC + i] = (_Float16)0.f;
  for (int i = tid; i < NPAD * 128; i += 256) sP[i] = (_Float16)0.f;

  // ---- Phase A: LayerNorm of the 100 window tokens into sH (f16) ----------
  for (int m = wvu; m < NTOK; m += 8) {
    const int grow = (wy * WIN + m / WIN) * WW + wx * WIN + (m % WIN);
    const float* xr = x + (size_t)grow * DIMC;
    float v[8], s = 0.f, ss = 0.f;
#pragma unroll
    for (int e = 0; e < 8; ++e) { v[e] = xr[lane * 8 + e]; s += v[e]; ss += v[e] * v[e]; }
#pragma unroll
    for (int off = 16; off; off >>= 1) {
      s  += __shfl_xor(s,  off, 32);
      ss += __shfl_xor(ss, off, 32);
    }
    const float mean = s * (1.f / 256.f);
    const float rstd = rsqrtf(ss * (1.f / 256.f) - mean * mean + 1e-5f);
#pragma unroll
    for (int e = 0; e < 8; ++e) {
      const int c = lane * 8 + e;
      sH[m * DIMC + c] = (_Float16)((v[e] - mean) * rstd * g1[c] + b1[c]);
    }
  }
  __syncthreads();

  // ---- Phase B: Q/K/V = LN(x) @ {wq, wkv} + bias  (WMMA, K=256) -----------
  {
    const int n = lane & 15, hh = lane >> 4;
    for (int mt = 0; mt < 7; ++mt) {
      v16h a[8];
#pragma unroll
      for (int ks = 0; ks < 8; ++ks)
        a[ks] = frag_a_rm(sH + mt * 16 * DIMC + ks * 32, DIMC);

      for (int nt = wvu; nt < 16; nt += 8) {          // ---- Q ----
        v8f c = {};
#pragma unroll
        for (int ks = 0; ks < 8; ++ks)
          c = wmma_f16(a[ks], frag_b_pk(wqP + ((size_t)ks * 16 + nt) * 512), c);
        const int col = nt * 16 + n;
        const float bi = bq[col];
#pragma unroll
        for (int r = 0; r < 8; ++r)
          sQ[(mt * 16 + 8 * hh + r) * DIMC + col] = (_Float16)((c[r] + bi) * SCALE_Q);
      }
      for (int nt = wvu; nt < 16; nt += 8) {          // ---- K ----
        v8f c = {};
#pragma unroll
        for (int ks = 0; ks < 8; ++ks)
          c = wmma_f16(a[ks], frag_b_pk(wkvP + ((size_t)ks * 32 + nt) * 512), c);
        const int col = nt * 16 + n;
        const float bi = bkv[col];
#pragma unroll
        for (int r = 0; r < 8; ++r)
          sK[(mt * 16 + 8 * hh + r) * DIMC + col] = (_Float16)(c[r] + bi);
      }
      for (int nt = wvu; nt < 16; nt += 8) {          // ---- V (transposed) --
        v8f c = {};
#pragma unroll
        for (int ks = 0; ks < 8; ++ks)
          c = wmma_f16(a[ks], frag_b_pk(wkvP + ((size_t)ks * 32 + 16 + nt) * 512), c);
        const int col = nt * 16 + n;
        const float bi = bkv[col + 256];
        v8h vv;
#pragma unroll
        for (int r = 0; r < 8; ++r) vv[r] = (_Float16)(c[r] + bi);
        *(v8h*)(sVT + col * 128 + mt * 16 + 8 * hh) = vv;   // one 16B store
      }
    }
  }
  __syncthreads();
  for (int i = tid; i < DIMC * 28; i += 256) {   // zero sVT token-pad 100..127
    const int c = i / 28, t = 100 + i % 28;
    sVT[c * 128 + t] = (_Float16)0.f;
  }
  __syncthreads();

  // ---- Phase C: per-head  S = QK^T (+bias) -> softmax -> O = P V ----------
  for (int h = 0; h < NH; ++h) {
    for (int job = wvu; job < 49; job += 8) {   // hd=32 => ONE wmma per tile
      const int mt = job / 7, nt = job % 7;
      v16h a = frag_a_rm(sQ + mt * 16 * DIMC + h * HD, DIMC);
      v16h b = frag_b_tr(sK + nt * 16 * DIMC + h * HD, DIMC);
      v8f c = {};
      c = wmma_f16(a, b, c);
      const int n = lane & 15, hh = lane >> 4;
      const int ncol = nt * 16 + n;
#pragma unroll
      for (int r = 0; r < 8; ++r) {
        const int m = mt * 16 + 8 * hh + r;
        float val = c[r];
        if (m < NTOK && ncol < NTOK) {
          const int di = m / WIN - ncol / WIN + (WIN - 1);
          const int dj = m % WIN - ncol % WIN + (WIN - 1);
          val += rpb[(di * (2 * WIN - 1) + dj) * NH + h];
        }
        sS[m * NPAD + ncol] = val;
      }
    }
    __syncthreads();

    for (int m = wvu; m < NTOK; m += 8) {   // softmax rows, wave32 shuffles
      const float* sr = sS + m * NPAD;
      float e0 = sr[lane];
      float e1 = sr[32 + lane];
      float e2 = sr[64 + lane];
      float e3 = (lane < 4) ? sr[96 + lane] : -3.0e38f;
      float mx = fmaxf(fmaxf(e0, e1), fmaxf(e2, e3));
#pragma unroll
      for (int off = 16; off; off >>= 1) mx = fmaxf(mx, __shfl_xor(mx, off, 32));
      e0 = __expf(e0 - mx);
      e1 = __expf(e1 - mx);
      e2 = __expf(e2 - mx);
      e3 = (lane < 4) ? __expf(e3 - mx) : 0.f;
      float sm = e0 + e1 + e2 + e3;
#pragma unroll
      for (int off = 16; off; off >>= 1) sm += __shfl_xor(sm, off, 32);
      const float inv = 1.f / sm;
      _Float16* pr = sP + m * 128;
      pr[lane]      = (_Float16)(e0 * inv);
      pr[32 + lane] = (_Float16)(e1 * inv);
      pr[64 + lane] = (_Float16)(e2 * inv);
      if (lane < 4) pr[96 + lane] = (_Float16)(e3 * inv);
    }
    __syncthreads();

    for (int job = wvu; job < 14; job += 8) {   // O_h = P @ V_h
      const int mt = job / 2, nt = job % 2;
      v8f c = {};
#pragma unroll
      for (int ks = 0; ks < 4; ++ks) {
        v16h a = frag_a_rm(sP + mt * 16 * 128 + ks * 32, 128);
        v16h b = frag_b_tr(sVT + (h * HD + nt * 16) * 128 + ks * 32, 128);
        c = wmma_f16(a, b, c);
      }
      const int n = lane & 15, hh = lane >> 4;
#pragma unroll
      for (int r = 0; r < 8; ++r) {
        const int m = mt * 16 + 8 * hh + r;
        sH[m * DIMC + h * HD + nt * 16 + n] = (_Float16)c[r];  // sH reused as O
      }
    }
    __syncthreads();
  }

  // ---- Phase D: X1 = x + O @ proj_w + proj_b  (A cached per mt) -----------
  {
    const int n = lane & 15, hh = lane >> 4;
    for (int mt = 0; mt < 7; ++mt) {
      v16h a[8];
#pragma unroll
      for (int ks = 0; ks < 8; ++ks)
        a[ks] = frag_a_rm(sH + mt * 16 * DIMC + ks * 32, DIMC);
      for (int nt = wvu; nt < 16; nt += 8) {
        v8f c = {};
#pragma unroll
        for (int ks = 0; ks < 8; ++ks)
          c = wmma_f16(a[ks], frag_b_pk(pwP + ((size_t)ks * 16 + nt) * 512), c);
        const int col = nt * 16 + n;
        const float bi = pb[col];
#pragma unroll
        for (int r = 0; r < 8; ++r) {
          const int m = mt * 16 + 8 * hh + r;
          if (m < NTOK) {
            const int grow = (wy * WIN + m / WIN) * WW + wx * WIN + (m % WIN);
            X1[(size_t)grow * DIMC + col] = x[(size_t)grow * DIMC + col] + c[r] + bi;
          }
        }
      }
    }
  }
}

// ===========================================================================
// Kernel 2: F = gelu(LN2(X1) @ l1_w + l1_b).  One block = 64 rows x ALL 1024
// cols: X1 read once, LN computed once; A cached in VGPRs per M-tile.
// ===========================================================================
__global__ __launch_bounds__(256)
void k_leff1(const float* __restrict__ X1,
             const float* __restrict__ g2, const float* __restrict__ b2,
             const _Float16* __restrict__ l1P, const float* __restrict__ l1b,
             _Float16* __restrict__ F) {
  __shared__ _Float16 sA[64 * DIMC];
  const int tid = threadIdx.x, lane = tid & 31;
  const int wvu = __builtin_amdgcn_readfirstlane(tid >> 5);
  const int m0 = blockIdx.x * 64;
  const int n = lane & 15, hh = lane >> 4;

  for (int r = wvu; r < 64; r += 8) {
    const float* xr = X1 + (size_t)(m0 + r) * DIMC;
    float v[8], s = 0.f, ss = 0.f;
#pragma unroll
    for (int e = 0; e < 8; ++e) { v[e] = xr[lane * 8 + e]; s += v[e]; ss += v[e] * v[e]; }
#pragma unroll
    for (int off = 16; off; off >>= 1) {
      s  += __shfl_xor(s,  off, 32);
      ss += __shfl_xor(ss, off, 32);
    }
    const float mean = s * (1.f / 256.f);
    const float rstd = rsqrtf(ss * (1.f / 256.f) - mean * mean + 1e-5f);
#pragma unroll
    for (int e = 0; e < 8; ++e) {
      const int c = lane * 8 + e;
      sA[r * DIMC + c] = (_Float16)((v[e] - mean) * rstd * g2[c] + b2[c]);
    }
  }
  __syncthreads();

  for (int mt = 0; mt < 4; ++mt) {
    v16h a[8];
#pragma unroll
    for (int ks = 0; ks < 8; ++ks)
      a[ks] = frag_a_rm(sA + mt * 16 * DIMC + ks * 32, DIMC);
    for (int nt = wvu; nt < 64; nt += 8) {            // 8 N-tiles per wave
      v8f c = {};
#pragma unroll
      for (int ks = 0; ks < 8; ++ks)
        c = wmma_f16(a[ks], frag_b_pk(l1P + ((size_t)ks * 64 + nt) * 512), c);
      const int col = nt * 16 + n;
      const float bi = l1b[col];
#pragma unroll
      for (int r = 0; r < 8; ++r) {
        const int m = m0 + mt * 16 + 8 * hh + r;
        F[(size_t)m * HID + col] = (_Float16)gelu_exact(c[r] + bi);
      }
    }
  }
}

// ===========================================================================
// Kernel 3: depthwise 3x3 conv + gelu, tiled: 8x8 pixels x 128-channel slab
// per block through LDS (halo 10x10xCS).  F read ~1.56x instead of 9x.
// ===========================================================================
__global__ __launch_bounds__(256)
void k_dwconv(const _Float16* __restrict__ F,
              const float* __restrict__ dww, const float* __restrict__ dwb,
              _Float16* __restrict__ G) {
  __shared__ _Float16 sF[100 * CS];      // 25600 B halo tile
  __shared__ float    sW[CS * 9];        // slab weights
  __shared__ float    sB[CS];            // slab bias
  const int tid = threadIdx.x;
  const int bx = blockIdx.x % 40, by = (blockIdx.x / 40) % 40;
  const int c0 = (blockIdx.x / 1600) * CS;
  const int pi0 = by * 8, pj0 = bx * 8;

  for (int i = tid; i < CS * 9; i += 256) sW[i] = dww[(size_t)(c0 + i / 9) * 9 + i % 9];
  for (int i = tid; i < CS; i += 256)     sB[i] = dwb[c0 + i];
  for (int idx = tid; idx < 100 * (CS / 8); idx += 256) {
    const int pos = idx / (CS / 8), ch8 = idx % (CS / 8);
    const int ii = pi0 - 1 + pos / 10, jj = pj0 - 1 + pos % 10;
    v8h val = {};
    if (ii >= 0 && ii < HH && jj >= 0 && jj < WW)
      val = *(const v8h*)(F + (size_t)(ii * WW + jj) * HID + c0 + ch8 * 8);
    *(v8h*)(sF + pos * CS + ch8 * 8) = val;
  }
  __syncthreads();

  const int pix = tid >> 2;              // 0..63 output pixel in tile
  const int ch0 = (tid & 3) * 32;        // 32-channel chunk
  const int oi = pix >> 3, oj = pix & 7;
#pragma unroll
  for (int g = 0; g < 4; ++g) {          // 4 x 8 channels
    const int cc = ch0 + g * 8;
    float acc[8];
#pragma unroll
    for (int e = 0; e < 8; ++e) acc[e] = sB[cc + e];
#pragma unroll
    for (int ky = 0; ky < 3; ++ky)
#pragma unroll
      for (int kx = 0; kx < 3; ++kx) {
        v8h t = *(const v8h*)(sF + ((oi + ky) * 10 + (oj + kx)) * CS + cc);
#pragma unroll
        for (int e = 0; e < 8; ++e)
          acc[e] += (float)t[e] * sW[(cc + e) * 9 + ky * 3 + kx];
      }
    v8h o;
#pragma unroll
    for (int e = 0; e < 8; ++e) o[e] = (_Float16)gelu_exact(acc[e]);
    *(v8h*)(G + (size_t)((pi0 + oi) * WW + pj0 + oj) * HID + c0 + cc) = o;
  }
}

// ===========================================================================
// Kernel 4: out = X1 + G @ l2_w + l2_b.  One block = 64 rows x ALL 256 cols:
// G read once.  k-outer, 64x32 A-slab staged in LDS; per k-step each wave
// holds 4 A frags + 2 B frags and issues 8 WMMAs into 8 live accumulators.
// ===========================================================================
__global__ __launch_bounds__(256)
void k_leff2(const _Float16* __restrict__ G,
             const _Float16* __restrict__ l2P, const float* __restrict__ l2b,
             const float* __restrict__ X1, float* __restrict__ out) {
  __shared__ _Float16 sA[64 * 32];
  const int tid = threadIdx.x, lane = tid & 31;
  const int wvu = __builtin_amdgcn_readfirstlane(tid >> 5);
  const int m0 = blockIdx.x * 64;
  const int nt0 = wvu * 2;                        // wave owns n-tiles nt0, nt0+1
  const int sr = tid >> 2, sc = (tid & 3) * 8;    // slab row / col segment

  v8f acc[8] = {};                                // [nt-half][mt]
  for (int ks = 0; ks < 32; ++ks) {
    const _Float16* src = G + (size_t)(m0 + sr) * HID + ks * 32 + sc;
    *(v8h*)(sA + sr * 32 + sc) = *(const v8h*)src;
    if (ks + 1 < 32) __builtin_prefetch(src + 32, 0, 1);
    __syncthreads();
    v16h a[4];
#pragma unroll
    for (int mt = 0; mt < 4; ++mt) a[mt] = frag_a_rm(sA + mt * 16 * 32, 32);
    v16h b0 = frag_b_pk(l2P + ((size_t)ks * 16 + nt0) * 512);
    v16h b1 = frag_b_pk(l2P + ((size_t)ks * 16 + nt0 + 1) * 512);
#pragma unroll
    for (int mt = 0; mt < 4; ++mt) {
      acc[mt]     = wmma_f16(a[mt], b0, acc[mt]);
      acc[4 + mt] = wmma_f16(a[mt], b1, acc[4 + mt]);
    }
    __syncthreads();
  }
  const int n = lane & 15, hh = lane >> 4;
#pragma unroll
  for (int j = 0; j < 8; ++j) {
    const int mt = j & 3, nt = nt0 + (j >> 2);
    const int col = nt * 16 + n;
    const float bi = l2b[col];
#pragma unroll
    for (int r = 0; r < 8; ++r) {
      const size_t m = m0 + mt * 16 + 8 * hh + r;
      out[m * DIMC + col] = X1[m * DIMC + col] + acc[j][r] + bi;
    }
  }
}

// ===========================================================================
extern "C" void kernel_launch(void* const* d_in, const int* in_sizes, int n_in,
                              void* d_out, int out_size, void* d_ws, size_t ws_size,
                              hipStream_t stream) {
  const float* x    = (const float*)d_in[0];
  const float* n1g  = (const float*)d_in[1];
  const float* n1b  = (const float*)d_in[2];
  const float* wq   = (const float*)d_in[3];
  const float* bq   = (const float*)d_in[4];
  const float* wkv  = (const float*)d_in[5];
  const float* bkv  = (const float*)d_in[6];
  const float* rpb  = (const float*)d_in[7];
  const float* pw   = (const float*)d_in[8];
  const float* pb   = (const float*)d_in[9];
  const float* n2g  = (const float*)d_in[10];
  const float* n2b  = (const float*)d_in[11];
  const float* l1w  = (const float*)d_in[12];
  const float* l1b  = (const float*)d_in[13];
  const float* dww  = (const float*)d_in[14];
  const float* dwb  = (const float*)d_in[15];
  const float* l2w  = (const float*)d_in[16];
  const float* l2b  = (const float*)d_in[17];

  const size_t TOK = (size_t)HH * WW;                 // 102400
  char* ws = (char*)d_ws;
  float*    X1 = (float*)ws;                          ws += TOK * DIMC * sizeof(float);
  _Float16* F  = (_Float16*)ws;                       ws += TOK * HID * sizeof(_Float16);
  _Float16* G  = (_Float16*)ws;                       ws += TOK * HID * sizeof(_Float16);
  _Float16* wqP  = (_Float16*)ws;                     ws += (size_t)DIMC * DIMC * 2;
  _Float16* wkvP = (_Float16*)ws;                     ws += (size_t)DIMC * 2 * DIMC * 2;
  _Float16* pwP  = (_Float16*)ws;                     ws += (size_t)DIMC * DIMC * 2;
  _Float16* l1P  = (_Float16*)ws;                     ws += (size_t)DIMC * HID * 2;
  _Float16* l2P  = (_Float16*)ws;                     ws += (size_t)HID * DIMC * 2;
  float*    out = (float*)d_out;

  // ---- pack weights into WMMA-B fragment layout (tiny, L2-resident) ------
  auto packgrid = [](int K, int N) { return ((K >> 5) * (N >> 4) * 32 + 255) / 256; };
  k_pack_b<<<packgrid(DIMC, DIMC), 256, 0, stream>>>(wq,  wqP,  DIMC, DIMC);
  k_pack_b<<<packgrid(DIMC, 2 * DIMC), 256, 0, stream>>>(wkv, wkvP, DIMC, 2 * DIMC);
  k_pack_b<<<packgrid(DIMC, DIMC), 256, 0, stream>>>(pw,  pwP,  DIMC, DIMC);
  k_pack_b<<<packgrid(DIMC, HID), 256, 0, stream>>>(l1w, l1P,  DIMC, HID);
  k_pack_b<<<packgrid(HID, DIMC), 256, 0, stream>>>(l2w, l2P,  HID, DIMC);

  // ~308 KB dynamic LDS for the attention kernel (CDNA5 WGP: 320 KB)
  const size_t lds_attn =
      (size_t)(NPAD * DIMC * 3 + DIMC * 128 + NPAD * 128) * sizeof(_Float16) +
      (size_t)NPAD * NPAD * sizeof(float);
  (void)hipFuncSetAttribute((const void*)k_win_attn,
                            hipFuncAttributeMaxDynamicSharedMemorySize,
                            (int)lds_attn);

  k_win_attn<<<dim3(1024), dim3(256), lds_attn, stream>>>(
      x, n1g, n1b, wqP, bq, wkvP, bkv, rpb, pwP, pb, X1);
  k_leff1<<<dim3((int)(TOK / 64)), dim3(256), 0, stream>>>(
      X1, n2g, n2b, l1P, l1b, F);
  k_dwconv<<<dim3(40 * 40 * (HID / CS)), dim3(256), 0, stream>>>(F, dww, dwb, G);
  k_leff2<<<dim3((int)(TOK / 64)), dim3(256), 0, stream>>>(
      G, l2P, l2b, X1, out);
}
